// GNN_87144886436622
// MI455X (gfx1250) — matrix operand order repaired and checked by reading, not compiled, not run on hardware
//
#include <hip/hip_runtime.h>
#include <hip/hip_bf16.h>
#include <math.h>

typedef __attribute__((ext_vector_type(16))) _Float16 v16h;
typedef __attribute__((ext_vector_type(8)))  _Float16 v8h;
typedef __attribute__((ext_vector_type(8)))  float    v8f;

#define LN_EPS 1e-5f

__device__ __forceinline__ v16h cat16(v8h lo, v8h hi) {
    v16h r;
#pragma unroll
    for (int i = 0; i < 8; ++i) { r[i] = lo[i]; r[i + 8] = hi[i]; }
    return r;
}

// =====================================================================
// Fused: optional LayerNorm -> fp16 convert -> y@W1 (and y@W2) via WMMA
// Block: 128 threads (4 waves), handles 64 node rows. D fixed at 64.
//   gamma==nullptr  -> no LayerNorm (final projection)
//   W2==nullptr     -> single GEMM
//   bias1!=nullptr  -> added to out1 columns
// =====================================================================
__global__ __launch_bounds__(128) void k_ln_gemm(
    const float* __restrict__ X,
    const float* __restrict__ gamma,
    const float* __restrict__ beta,
    const float* __restrict__ W1,
    const float* __restrict__ W2,
    const float* __restrict__ bias1,
    float* __restrict__ out1,
    float* __restrict__ out2,
    int nrows)
{
    __shared__ __align__(16) _Float16 y16[64][64];   // normalized input tile (fp16)
    __shared__ __align__(16) _Float16 w1t[64][64];   // W1 transposed: [n][k]
    __shared__ __align__(16) _Float16 w2t[64][64];   // W2 transposed: [n][k]
    __shared__ float redS[64][2];
    __shared__ float redQ[64][2];

    const int t = threadIdx.x;
    const int blockRow = blockIdx.x * 64;

    // ---------- phase 1: load 64x64 tile, LayerNorm (optional), fp16 ----------
    {
        const int r    = t & 63;   // row in tile
        const int half = t >> 6;   // 0/1: which 32 features
        const int node = blockRow + r;
        float4 buf[8];
        if (node < nrows) {
            const float4* rp = (const float4*)(X + (size_t)node * 64) + half * 8;
#pragma unroll
            for (int q = 0; q < 8; ++q) buf[q] = rp[q];
        } else {
#pragma unroll
            for (int q = 0; q < 8; ++q) buf[q] = make_float4(0.f, 0.f, 0.f, 0.f);
        }
        if (gamma) {
            float s = 0.f, ss = 0.f;
#pragma unroll
            for (int q = 0; q < 8; ++q) {
                s  += buf[q].x + buf[q].y + buf[q].z + buf[q].w;
                ss += buf[q].x * buf[q].x + buf[q].y * buf[q].y
                    + buf[q].z * buf[q].z + buf[q].w * buf[q].w;
            }
            redS[r][half] = s; redQ[r][half] = ss;
            __syncthreads();
            const float sum  = redS[r][0] + redS[r][1];
            const float sq   = redQ[r][0] + redQ[r][1];
            const float mean = sum * (1.f / 64.f);
            const float var  = sq * (1.f / 64.f) - mean * mean;
            const float rstd = rsqrtf(var + LN_EPS);
#pragma unroll
            for (int q = 0; q < 8; ++q) {
                const int d = half * 32 + q * 4;
                const float4 g = ((const float4*)gamma)[d >> 2];
                const float4 b = ((const float4*)beta)[d >> 2];
                y16[r][d + 0] = (_Float16)((buf[q].x - mean) * rstd * g.x + b.x);
                y16[r][d + 1] = (_Float16)((buf[q].y - mean) * rstd * g.y + b.y);
                y16[r][d + 2] = (_Float16)((buf[q].z - mean) * rstd * g.z + b.z);
                y16[r][d + 3] = (_Float16)((buf[q].w - mean) * rstd * g.w + b.w);
            }
        } else {
#pragma unroll
            for (int q = 0; q < 8; ++q) {
                const int d = half * 32 + q * 4;
                y16[r][d + 0] = (_Float16)buf[q].x;
                y16[r][d + 1] = (_Float16)buf[q].y;
                y16[r][d + 2] = (_Float16)buf[q].z;
                y16[r][d + 3] = (_Float16)buf[q].w;
            }
        }
    }

    // ---------- phase 2: weights -> LDS, transposed, fp16 ----------
    for (int idx = t; idx < 4096; idx += 128) {
        const int k = idx >> 6, n = idx & 63;
        w1t[n][k] = (_Float16)W1[idx];
        if (W2) w2t[n][k] = (_Float16)W2[idx];
    }
    __syncthreads();

    // ---------- phase 3: WMMA. wave w owns rows [w*16, w*16+16) ----------
    const int wv   = t >> 5;
    const int lane = t & 31;
    const int mloc = lane & 15;
    const int hi16 = lane >> 4;      // 0 for lanes 0-15, 1 for lanes 16-31
    const int mrow = wv * 16 + mloc;
    const int klo  = hi16 * 8;       // unused; A frags also use kb
    const int kb   = hi16 * 16;      // B-frag K sub-base within a 32-K step

    // A fragments (16x32 f16 layout): lanes 0-15 hold K[0-15]; 16-31 hold K[16-31]
    const _Float16* ap = &y16[mrow][0];
    const v16h a0 = cat16(*(const v8h*)(ap + kb),      *(const v8h*)(ap + kb + 8));
    const v16h a1 = cat16(*(const v8h*)(ap + kb + 32), *(const v8h*)(ap + kb + 40));

#pragma unroll
    for (int nt = 0; nt < 4; ++nt) {
        const int ncol = nt * 16 + mloc;
        const _Float16* b1p = &w1t[ncol][0];
        const v16h b10 = cat16(*(const v8h*)(b1p + kb),      *(const v8h*)(b1p + kb + 8));
        const v16h b11 = cat16(*(const v8h*)(b1p + kb + 32), *(const v8h*)(b1p + kb + 40));
        v8f c1 = {};
        c1 = __builtin_amdgcn_wmma_f32_16x16x32_f16(false, a0, false, b10, (short)0, c1, false, false);
        c1 = __builtin_amdgcn_wmma_f32_16x16x32_f16(false, a1, false, b11, (short)0, c1, false, false);
        v8f c2 = {};
        if (W2) {
            const _Float16* b2p = &w2t[ncol][0];
            const v16h b20 = cat16(*(const v8h*)(b2p + kb),      *(const v8h*)(b2p + kb + 8));
            const v16h b21 = cat16(*(const v8h*)(b2p + kb + 32), *(const v8h*)(b2p + kb + 40));
            c2 = __builtin_amdgcn_wmma_f32_16x16x32_f16(false, a0, false, b20, (short)0, c2, false, false);
            c2 = __builtin_amdgcn_wmma_f32_16x16x32_f16(false, a1, false, b21, (short)0, c2, false, false);
        }
        const float badd = bias1 ? bias1[ncol] : 0.f;
        // C/D layout: VGPR v -> row (hi16*8 + v), col = lane%16
#pragma unroll
        for (int v = 0; v < 8; ++v) {
            const int m = blockRow + wv * 16 + hi16 * 8 + v;
            if (m < nrows) {
                out1[(size_t)m * 64 + ncol] = c1[v] + badd;
                if (W2) out2[(size_t)m * 64 + ncol] = c2[v];
            }
        }
    }
}

// =====================================================================
// CSR build (once per call; graph is static across layers)
// =====================================================================
__global__ void k_zero_int(int* __restrict__ p, int n) {
    const int i = blockIdx.x * blockDim.x + threadIdx.x;
    if (i < n) p[i] = 0;
}

__global__ void k_count(const int* __restrict__ edge, int Ee, int nrows,
                        int* __restrict__ deg) {
    const int j = blockIdx.x * blockDim.x + threadIdx.x;
    const int Et = Ee + nrows;
    if (j >= Et) return;
    const int tg = (j < Ee) ? edge[Ee + j] : (j - Ee);
    atomicAdd(&deg[tg], 1);
}

// single-block exclusive scan: deg[0..n) -> rowptr[0..n]
__global__ __launch_bounds__(1024) void k_scan(const int* __restrict__ deg,
                                               int* __restrict__ rowptr, int n) {
    __shared__ int part[1024];
    const int tid = threadIdx.x;
    const int chunk = (n + 1023) / 1024;
    const int lo = tid * chunk;
    const int hi = min(lo + chunk, n);
    int s = 0;
    for (int i = lo; i < hi; ++i) s += deg[i];
    part[tid] = s;
    __syncthreads();
    if (tid == 0) {
        int run = 0;
        for (int i = 0; i < 1024; ++i) { const int v = part[i]; part[i] = run; run += v; }
    }
    __syncthreads();
    int run = part[tid];
    for (int i = lo; i < hi; ++i) { rowptr[i] = run; run += deg[i]; }
    if (lo < n && hi == n) rowptr[n] = run;   // total
}

__global__ void k_fill(const int* __restrict__ edge, int Ee, int nrows,
                       const int* __restrict__ rowptr, int* __restrict__ cur,
                       int* __restrict__ csrc) {
    const int j = blockIdx.x * blockDim.x + threadIdx.x;
    const int Et = Ee + nrows;
    if (j >= Et) return;
    int s, tg;
    if (j < Ee) { s = edge[j]; tg = edge[Ee + j]; } else { s = j - Ee; tg = s; }
    const int p = rowptr[tg] + atomicAdd(&cur[tg], 1);
    csrc[p] = s;
}

// sort each node's in-edge segment -> deterministic summation order
// (duplicate (s,t) edges contribute identically, so sorted order is bit-stable)
__global__ void k_sortseg(const int* __restrict__ rowptr, int* __restrict__ csrc,
                          int nrows) {
    const int node = blockIdx.x * blockDim.x + threadIdx.x;
    if (node >= nrows) return;
    const int s = rowptr[node], e = rowptr[node + 1];
    for (int i = s + 1; i < e; ++i) {
        const int v = csrc[i];
        int k = i - 1;
        while (k >= s && csrc[k] > v) { csrc[k + 1] = csrc[k]; --k; }
        csrc[k + 1] = v;
    }
}

// =====================================================================
// One wave per target node: single-pass online-softmax GATv2 aggregation,
// fused with conv bias + exact GELU + residual into h. Zero atomics.
// Lane l owns feature dims {l, l+32} (coalesced 128B row accesses).
// =====================================================================
__global__ __launch_bounds__(256) void k_gat_node(
    const float* __restrict__ xl, const float* __restrict__ xr,
    const float* __restrict__ att, const float* __restrict__ cbias,
    const int* __restrict__ rowptr, const int* __restrict__ csrc,
    float* __restrict__ h, int nrows)
{
    const int gtid = blockIdx.x * blockDim.x + threadIdx.x;
    const int node = gtid >> 5;
    const int lane = gtid & 31;
    if (node >= nrows) return;

    const float a0  = att[lane];
    const float a1  = att[lane + 32];
    const float xr0 = xr[(size_t)node * 64 + lane];
    const float xr1 = xr[(size_t)node * 64 + 32 + lane];

    const int start = rowptr[node], end = rowptr[node + 1];
    float m = -INFINITY, den = 0.f, acc0 = 0.f, acc1 = 0.f;

    for (int j = start; j < end; ++j) {
        const int s = csrc[j];
        const float x0 = xl[(size_t)s * 64 + lane];
        const float x1 = xl[(size_t)s * 64 + 32 + lane];
        float t0 = x0 + xr0; t0 = t0 > 0.f ? t0 : 0.2f * t0;
        float t1 = x1 + xr1; t1 = t1 > 0.f ? t1 : 0.2f * t1;
        float p = a0 * t0 + a1 * t1;
        // wave32 butterfly reduce -> all lanes hold full dot product
        p += __shfl_xor(p, 16);
        p += __shfl_xor(p, 8);
        p += __shfl_xor(p, 4);
        p += __shfl_xor(p, 2);
        p += __shfl_xor(p, 1);
        const float e    = p;
        const float nm   = fmaxf(m, e);
        const float corr = expf(m - nm);   // 0 on first edge (m = -inf)
        const float w    = expf(e - nm);
        den  = den  * corr + w;
        acc0 = acc0 * corr + w * x0;
        acc1 = acc1 * corr + w * x1;
        m = nm;
    }

    const float inv = 1.f / (den + 1e-16f);
    float o0 = acc0 * inv + cbias[lane];
    float o1 = acc1 * inv + cbias[lane + 32];
    o0 = 0.5f * o0 * (1.f + erff(o0 * 0.70710678118654752f));
    o1 = 0.5f * o1 * (1.f + erff(o1 * 0.70710678118654752f));
    h[(size_t)node * 64 + lane]      += o0;
    h[(size_t)node * 64 + 32 + lane] += o1;
}

extern "C" void kernel_launch(void* const* d_in, const int* in_sizes, int n_in,
                              void* d_out, int out_size, void* d_ws, size_t ws_size,
                              hipStream_t stream) {
    // input order per setup_inputs():
    // 0:x(unused) 1:edge_index 2:emb_weight 3:ln_gamma 4:ln_beta 5:Wl 6:Wr 7:att 8:conv_bias 9:post_w 10:post_b
    const int*   edge  = (const int*)d_in[1];
    const float* emb   = (const float*)d_in[2];
    const float* ln_g  = (const float*)d_in[3];
    const float* ln_b  = (const float*)d_in[4];
    const float* Wl    = (const float*)d_in[5];
    const float* Wr    = (const float*)d_in[6];
    const float* att   = (const float*)d_in[7];
    const float* cbias = (const float*)d_in[8];
    const float* postw = (const float*)d_in[9];
    const float* postb = (const float*)d_in[10];

    const int Nn = in_sizes[2] / 64;   // 100000
    const int Ee = in_sizes[1] / 2;    // 1200000
    const int Et = Ee + Nn;
    const int L  = in_sizes[3] / 64;   // 2

    // workspace layout
    float* h      = (float*)d_ws;
    float* xl     = h  + (size_t)Nn * 64;
    float* xr     = xl + (size_t)Nn * 64;
    int*   rowptr = (int*)(xr + (size_t)Nn * 64);   // Nn+1
    int*   deg    = rowptr + (Nn + 1);              // Nn (reused as fill cursor)
    int*   csrc   = deg + Nn;                       // Et

    hipMemcpyAsync(h, emb, (size_t)Nn * 64 * sizeof(float),
                   hipMemcpyDeviceToDevice, stream);

    const int gemm_blocks = (Nn + 63) / 64;
    const int ew_blocks   = (Et + 255) / 256;
    const int n_blocks    = (Nn + 255) / 256;
    const int nw_blocks   = ((Nn * 32) + 255) / 256;   // wave per node

    // ---- build CSR (by target) once; deterministic after per-segment sort ----
    k_zero_int<<<n_blocks, 256, 0, stream>>>(deg, Nn);
    k_count<<<ew_blocks, 256, 0, stream>>>(edge, Ee, Nn, deg);
    k_scan<<<1, 1024, 0, stream>>>(deg, rowptr, Nn);
    k_zero_int<<<n_blocks, 256, 0, stream>>>(deg, Nn);
    k_fill<<<ew_blocks, 256, 0, stream>>>(edge, Ee, Nn, rowptr, deg, csrc);
    k_sortseg<<<n_blocks, 256, 0, stream>>>(rowptr, csrc, Nn);

    for (int i = 0; i < L; ++i) {
        k_ln_gemm<<<gemm_blocks, 128, 0, stream>>>(
            h, ln_g + (size_t)i * 64, ln_b + (size_t)i * 64,
            Wl + (size_t)i * 4096, Wr + (size_t)i * 4096,
            nullptr, xl, xr, Nn);
        k_gat_node<<<nw_blocks, 256, 0, stream>>>(
            xl, xr, att + (size_t)i * 64, cbias + (size_t)i * 64,
            rowptr, csrc, h, Nn);
    }

    // final projection: out = h @ post_w + post_b
    k_ln_gemm<<<gemm_blocks, 128, 0, stream>>>(
        h, nullptr, nullptr, postw, nullptr, postb, (float*)d_out, nullptr, Nn);
}
